// MultiHeadAttention_27410481283623
// MI455X (gfx1250) — matrix operand order repaired and checked by reading, not compiled, not run on hardware
//
#include <hip/hip_runtime.h>
#include <hip/hip_bf16.h>

// ---------------------------------------------------------------------------
// MI455X (gfx1250) LoRA multi-head attention, bf16 WMMA pipeline.
// ---------------------------------------------------------------------------

typedef __bf16 bf16_t;
typedef __attribute__((ext_vector_type(16))) __bf16 v16bf;
typedef __attribute__((ext_vector_type(8)))  float  v8f;

#define DD    2048   // model dim (= K = N of every GEMM)
#define BSZ   2      // batch
#define SEQ   2048   // sequence
#define HEADS 16
#define DKH   128    // head dim
#define MROWS (BSZ*SEQ)   // 4096 GEMM rows
#define LRANK 8

__device__ __forceinline__ v8f wmma_bf16(v16bf a, v16bf b, v8f c) {
  // D(16x16 f32) = A(16x32 bf16) * B(32x16 bf16) + C
  return __builtin_amdgcn_wmma_f32_16x16x32_bf16(
      /*neg_a=*/false, a, /*neg_b=*/false, b,
      /*c_mod=*/(short)0, c, /*reuse_a=*/false, /*reuse_b=*/false);
}

// ---------------------------------------------------------------------------
// 1) Weff^T[n][k] = (W[k][n] + sum_r la[k][r]*lb[r][n]) in bf16.
//    32x32 LDS tile transpose: coalesced reads AND writes.
// ---------------------------------------------------------------------------
__global__ void prep_weight_kernel(const float* __restrict__ w,
                                   const float* __restrict__ la,
                                   const float* __restrict__ lb,
                                   bf16_t* __restrict__ wt) {
  __shared__ bf16_t tile[32][33];
  const int n0 = blockIdx.x * 32, k0 = blockIdx.y * 32;
  const int tx = threadIdx.x & 31, ty = threadIdx.x >> 5;   // 32 x 8
#pragma unroll
  for (int p = 0; p < 4; ++p) {
    const int k = k0 + ty + p * 8;
    const int n = n0 + tx;
    float acc = w[(size_t)k * DD + n];
#pragma unroll
    for (int r = 0; r < LRANK; ++r)
      acc += la[(size_t)k * LRANK + r] * lb[(size_t)r * DD + n];
    tile[ty + p * 8][tx] = (bf16_t)acc;
  }
  __syncthreads();
#pragma unroll
  for (int p = 0; p < 4; ++p) {
    const int row = ty + p * 8;                  // local n
    wt[(size_t)(n0 + row) * DD + k0 + tx] = tile[tx][row];
  }
}

// ---------------------------------------------------------------------------
// 2) GEMM: Out[M=4096, N=2048] = X[M,2048] @ Weff[2048,N] + bias
//    Weff passed transposed (Wt[n][k], bf16). Block = 256 thr = 8 waves,
//    macro tile 128(M) x 64(N); wave tile 32x32 = 2x2 WMMA accums.
//    OUT_MODE: 0 = fp32 [M,N] (final output)
//              1 = bf16 [B,H,S,dk]  (Q, K)
//              2 = bf16 [B,H,dk,S]  (V, pre-transposed for flash P@V)
// ---------------------------------------------------------------------------
template <bool IN_F32, int OUT_MODE>
__global__ __launch_bounds__(256, 1)
void gemm_bf16_wmma(const void* __restrict__ Xv,
                    const bf16_t* __restrict__ Wt,
                    const float* __restrict__ bias,
                    void* __restrict__ Outv) {
  __shared__ bf16_t xs[128][40];                 // padded: conflict-free A reads
  const int t    = threadIdx.x;
  const int lane = t & 31, wv = t >> 5;
  const int wm = wv & 3, wn = wv >> 2;           // 4 (M) x 2 (N) waves
  const int m0 = blockIdx.y * 128;
  const int n0 = blockIdx.x * 64;
  const int l16 = lane & 15, lhalf = lane >> 4;
  const int khalf8 = lhalf * 8, khalf16 = lhalf * 16;

  v8f acc[2][2] = {};

  const int lr = t >> 1;                          // staging: row 0..127
  const int lc = (t & 1) * 16;                    // col 0 / 16

  for (int k0 = 0; k0 < DD; k0 += 32) {
    // ---- stage 128x32 X tile into LDS (fp32 -> bf16 on the fly) ----
    if (IN_F32) {
      const float* xr = (const float*)Xv + (size_t)(m0 + lr) * DD + k0 + lc;
      if (k0 + 32 < DD) __builtin_prefetch(xr + 32, 0, 3);
#pragma unroll
      for (int i = 0; i < 16; ++i) xs[lr][lc + i] = (bf16_t)xr[i];
    } else {
      const bf16_t* xr = (const bf16_t*)Xv + (size_t)(m0 + lr) * DD + k0 + lc;
      if (k0 + 32 < DD) __builtin_prefetch(xr + 32, 0, 3);
#pragma unroll
      for (int i = 0; i < 16; ++i) xs[lr][lc + i] = xr[i];
    }
    __syncthreads();

    // ---- A fragments (16x32) from LDS: one contiguous 32B run per lane ----
    v16bf a[2];
#pragma unroll
    for (int mt = 0; mt < 2; ++mt) {
      const bf16_t* ar = &xs[wm * 32 + mt * 16 + l16][0];
#pragma unroll
      for (int i = 0; i < 8; ++i) {
        a[mt][i]     = ar[khalf16 + i];
        a[mt][i + 8] = ar[khalf16 + 8 + i];
      }
    }
    // ---- B fragments straight from L2-resident Wt (contiguous 32B/lane) ----
#pragma unroll
    for (int nt = 0; nt < 2; ++nt) {
      const int n = n0 + wn * 32 + nt * 16 + l16;
      v16bf bfr = *(const v16bf*)&Wt[(size_t)n * DD + k0 + khalf16];
#pragma unroll
      for (int mt = 0; mt < 2; ++mt)
        acc[mt][nt] = wmma_bf16(a[mt], bfr, acc[mt][nt]);
    }
    __syncthreads();
  }

  // ---- epilogue: bias + layout-specific store ----
#pragma unroll
  for (int mt = 0; mt < 2; ++mt) {
#pragma unroll
    for (int nt = 0; nt < 2; ++nt) {
      const int n  = n0 + wn * 32 + nt * 16 + l16;
      const float bv = bias[n];
#pragma unroll
      for (int r = 0; r < 8; ++r) {
        const int m = m0 + wm * 32 + mt * 16 + r + lhalf * 8;
        const float v = acc[mt][nt][r] + bv;
        if (OUT_MODE == 0) {
          ((float*)Outv)[(size_t)m * DD + n] = v;
        } else {
          const int b = m >> 11, s = m & (SEQ - 1);
          const int h = n >> 7,  d = n & (DKH - 1);
          if (OUT_MODE == 1)
            ((bf16_t*)Outv)[(((size_t)b * HEADS + h) * SEQ + s) * DKH + d] = (bf16_t)v;
          else
            ((bf16_t*)Outv)[(((size_t)b * HEADS + h) * DKH + d) * SEQ + s] = (bf16_t)v;
        }
      }
    }
  }
}

// ---------------------------------------------------------------------------
// 3) Flash attention: one wave owns a 16-row Q tile; streams 32-key tiles.
//    S = Q@K^T via 2x4 WMMAs, online softmax (half-wave shuffle reductions
//    match the f32 C layout: one row == one VGPR across 16 lanes), P is
//    bounced through per-wave LDS to re-shape C-layout -> A-layout, then
//    O += P@V via 8 WMMAs against pre-transposed V (contiguous B-frags).
//    __launch_bounds__(256,1): ~160 live VGPRs/wave -> no scratch spills.
// ---------------------------------------------------------------------------
__global__ __launch_bounds__(256, 1)
void flash_attn_kernel(const bf16_t* __restrict__ Q,
                       const bf16_t* __restrict__ Kt,
                       const bf16_t* __restrict__ Vt,
                       bf16_t* __restrict__ Ctx) {
  __shared__ bf16_t pbuf[8][16][40];
  const int t = threadIdx.x, lane = t & 31, wv = t >> 5;
  const int gw = blockIdx.x * 8 + wv;            // global wave id
  const int qt = gw & 127;                       // S/16 = 128 q-tiles
  const int h  = (gw >> 7) & (HEADS - 1);
  const int b  = gw >> 11;
  const int l16 = lane & 15, lhalf = lane >> 4;
  const int khalf8 = lhalf * 8, khalf16 = lhalf * 16;
  const float scale = 0.08838834764831845f;      // 1/sqrt(128)

  const size_t bh = (size_t)b * HEADS + h;

  // Q tile as 4 A-fragments (dk = 128 = 4 x 32), kept register-resident
  const bf16_t* qbase = Q + (bh * SEQ + qt * 16 + l16) * DKH;
  v16bf qa[4];
#pragma unroll
  for (int c = 0; c < 4; ++c) {
#pragma unroll
    for (int i = 0; i < 8; ++i) {
      qa[c][i]     = qbase[c * 32 + khalf16 + i];
      qa[c][i + 8] = qbase[c * 32 + khalf16 + 8 + i];
    }
  }

  v8f o[8] = {};                                  // 16 x 128 fp32 accumulator
  float mi[8], li[8];
#pragma unroll
  for (int r = 0; r < 8; ++r) { mi[r] = -3.0e38f; li[r] = 0.f; }

  const int ntiles = (qt >> 1) + 1;               // causal: keys <= qt*16+15
  for (int kt = 0; kt < ntiles; ++kt) {
    const int kb = kt * 32;

    // ---- S = Q @ K^T (16 x 32 tile as two 16x16 C-frags) ----
    v8f s[2] = {};
#pragma unroll
    for (int nt = 0; nt < 2; ++nt) {
      const bf16_t* krow = Kt + (bh * SEQ + kb + nt * 16 + l16) * DKH;
#pragma unroll
      for (int c = 0; c < 4; ++c) {
        v16bf bfr = *(const v16bf*)(krow + c * 32 + khalf16);
        s[nt] = wmma_bf16(qa[c], bfr, s[nt]);
      }
    }

    // ---- causal mask + online softmax (rows live in half-waves) ----
    const int qrow0 = qt * 16 + lhalf * 8;
#pragma unroll
    for (int r = 0; r < 8; ++r) {
      const int qr = qrow0 + r;
      float v0 = s[0][r] * scale;
      float v1 = s[1][r] * scale;
      if (kb + l16 > qr)      v0 = -3.0e38f;
      if (kb + 16 + l16 > qr) v1 = -3.0e38f;
      float mx = fmaxf(v0, v1);
#pragma unroll
      for (int j = 8; j >= 1; j >>= 1) mx = fmaxf(mx, __shfl_xor(mx, j, 32));
      const float mnew  = fmaxf(mi[r], mx);
      const float alpha = __expf(mi[r] - mnew);
      const float p0 = __expf(v0 - mnew);
      const float p1 = __expf(v1 - mnew);
      float rs = p0 + p1;
#pragma unroll
      for (int j = 8; j >= 1; j >>= 1) rs += __shfl_xor(rs, j, 32);
      li[r] = li[r] * alpha + rs;
      mi[r] = mnew;
#pragma unroll
      for (int cc = 0; cc < 8; ++cc) o[cc][r] *= alpha;
      const int lrow = r + lhalf * 8;
      pbuf[wv][lrow][l16]      = (bf16_t)p0;     // C-layout -> LDS
      pbuf[wv][lrow][16 + l16] = (bf16_t)p1;
    }

    asm volatile("s_wait_dscnt 0" ::: "memory");  // LDS RAW across lanes

    // ---- re-read P in A-fragment layout ----
    v16bf pa;
#pragma unroll
    for (int i = 0; i < 8; ++i) {
      pa[i]     = pbuf[wv][l16][khalf16 + i];
      pa[i + 8] = pbuf[wv][l16][khalf16 + 8 + i];
    }

    // ---- O += P @ V (V pre-transposed: contiguous 32B B-frag loads) ----
#pragma unroll
    for (int cc = 0; cc < 8; ++cc) {
      const bf16_t* vrow = Vt + (bh * DKH + cc * 16 + l16) * SEQ;
      v16bf bfr = *(const v16bf*)(vrow + kb + khalf16);
      o[cc] = wmma_bf16(pa, bfr, o[cc]);
    }
  }

  // ---- normalize and store context bf16 [B,S,D] ----
#pragma unroll
  for (int r = 0; r < 8; ++r) {
    const float inv = 1.0f / li[r];
    const int qrow = qt * 16 + r + lhalf * 8;
    bf16_t* crow = Ctx + ((size_t)b * SEQ + qrow) * DD + h * DKH;
#pragma unroll
    for (int cc = 0; cc < 8; ++cc)
      crow[cc * 16 + l16] = (bf16_t)(o[cc][r] * inv);
  }
}

// ---------------------------------------------------------------------------
// Launch
// ---------------------------------------------------------------------------
extern "C" void kernel_launch(void* const* d_in, const int* in_sizes, int n_in,
                              void* d_out, int out_size, void* d_ws, size_t ws_size,
                              hipStream_t stream) {
  (void)in_sizes; (void)n_in; (void)out_size; (void)ws_size;
  const float* query = (const float*)d_in[0];
  const float* keyin = (const float*)d_in[1];
  const float* value = (const float*)d_in[2];
  const float* wq = (const float*)d_in[3],  *bq = (const float*)d_in[4];
  const float* laq = (const float*)d_in[5], *lbq = (const float*)d_in[6];
  const float* wk = (const float*)d_in[7],  *bk = (const float*)d_in[8];
  const float* lak = (const float*)d_in[9], *lbk = (const float*)d_in[10];
  const float* wv = (const float*)d_in[11], *bv = (const float*)d_in[12];
  const float* lav = (const float*)d_in[13],*lbv = (const float*)d_in[14];
  const float* wo = (const float*)d_in[15], *bo = (const float*)d_in[16];
  const float* lao = (const float*)d_in[17],*lbo = (const float*)d_in[18];

  // workspace carve-up (bf16 everywhere)
  const size_t WTE  = (size_t)DD * DD;            // 4.19M elems (8 MB)
  const size_t ACTE = (size_t)BSZ * SEQ * DD;     // 8.39M elems (16 MB)
  char* ws = (char*)d_ws;
  bf16_t* wtq = (bf16_t*)ws;                      ws += WTE * 2;
  bf16_t* wtk = (bf16_t*)ws;                      ws += WTE * 2;
  bf16_t* wtv = (bf16_t*)ws;                      ws += WTE * 2;
  bf16_t* wto = (bf16_t*)ws;                      ws += WTE * 2;
  bf16_t* Qb  = (bf16_t*)ws;                      ws += ACTE * 2;  // [B,H,S,dk]
  bf16_t* Kb  = (bf16_t*)ws;                      ws += ACTE * 2;  // [B,H,S,dk]
  bf16_t* Vtb = (bf16_t*)ws;                      ws += ACTE * 2;  // [B,H,dk,S]
  bf16_t* Ctx = (bf16_t*)ws;                      ws += ACTE * 2;  // [B,S,D]

  const dim3 blk(256);
  const dim3 pgrid(DD / 32, DD / 32);             // 64 x 64
  prep_weight_kernel<<<pgrid, blk, 0, stream>>>(wq, laq, lbq, wtq);
  prep_weight_kernel<<<pgrid, blk, 0, stream>>>(wk, lak, lbk, wtk);
  prep_weight_kernel<<<pgrid, blk, 0, stream>>>(wv, lav, lbv, wtv);
  prep_weight_kernel<<<pgrid, blk, 0, stream>>>(wo, lao, lbo, wto);

  const dim3 ggrid(DD / 64, MROWS / 128);         // 32 x 32
  gemm_bf16_wmma<true, 1><<<ggrid, blk, 0, stream>>>(query, wtq, bq, Qb);
  gemm_bf16_wmma<true, 1><<<ggrid, blk, 0, stream>>>(keyin, wtk, bk, Kb);
  gemm_bf16_wmma<true, 2><<<ggrid, blk, 0, stream>>>(value, wtv, bv, Vtb);

  flash_attn_kernel<<<(BSZ * HEADS * (SEQ / 16)) / 8, blk, 0, stream>>>(Qb, Kb, Vtb, Ctx);

  gemm_bf16_wmma<false, 0><<<ggrid, blk, 0, stream>>>(Ctx, wto, bo, d_out);
}
